// CrossAttention_22900765623037
// MI455X (gfx1250) — matrix operand order repaired and checked by reading, not compiled, not run on hardware
//
#include <hip/hip_runtime.h>
#include <hip/hip_bf16.h>

typedef __attribute__((ext_vector_type(16))) __bf16 v16bf;
typedef __attribute__((ext_vector_type(8)))  float  v8f;

#define BB     2
#define NN     2048
#define MMK    2048
#define QD     1024
#define HH     16
#define DH     64
#define DMODEL 1024
#define SCALEF 0.125f   // 64^-0.5

// ---------- helpers ----------

__device__ __forceinline__ __bf16 f2bf(float f) {
  unsigned u = __builtin_bit_cast(unsigned, f);
  unsigned r = u + 0x7FFFu + ((u >> 16) & 1u);   // round-to-nearest-even
  unsigned short h = (unsigned short)(r >> 16);
  return __builtin_bit_cast(__bf16, h);
}

__device__ __forceinline__ v8f wmma_bf16(v16bf a, v16bf b, v8f c) {
  // D = A(16x32 bf16) * B(32x16 bf16) + C(16x16 f32)
  return __builtin_amdgcn_wmma_f32_16x16x32_bf16(
      /*neg_a=*/false, a, /*neg_b=*/false, b,
      /*c_mod=*/(short)0, c, /*reuse_a=*/false, /*reuse_b=*/false);
}

// A fragment (16x32, row-major src, ISA 7.12.2 layout):
// lane m=ln&15, half=ln>>4 holds k = {8*half+0..7, 16+8*half+0..7} of row m.
__device__ __forceinline__ v16bf load_frag_a(const __bf16* base, int ld, int row0, int k0) {
  const int ln = threadIdx.x & 31;
  const int m = ln & 15, half = ln >> 4;
  const __bf16* p = base + (size_t)(row0 + m) * ld + k0 + 8 * half;
  union { v16bf v; uint4 q[2]; } u;
  u.q[0] = *(const uint4*)p;
  u.q[1] = *(const uint4*)(p + 16);
  return u.v;
}

// B fragment (32x16): B[k][n] = W[n0+n][k0+k], lane n=ln&15, half=ln>>4
// holds k = {8*half+0..7, 16+8*half+0..7} of column n -> two 16B loads per lane.
__device__ __forceinline__ v16bf load_frag_b(const __bf16* base, int ld, int n0, int k0) {
  const int ln = threadIdx.x & 31;
  const int n = ln & 15, half = ln >> 4;
  const __bf16* p = base + (size_t)(n0 + n) * ld + k0 + 8 * half;
  union { v16bf v; uint4 q[2]; } u;
  u.q[0] = *(const uint4*)p;
  u.q[1] = *(const uint4*)(p + 16);
  return u.v;
}

// A fragment from a 16x32 bf16 tile in LDS (row stride 32 elements)
__device__ __forceinline__ v16bf load_frag_a_lds(const __bf16* lds) {
  const int ln = threadIdx.x & 31;
  const int m = ln & 15, half = ln >> 4;
  const __bf16* p = lds + m * 32 + 8 * half;
  union { v16bf v; uint4 q[2]; } u;
  u.q[0] = *(const uint4*)p;
  u.q[1] = *(const uint4*)(p + 16);
  return u.v;
}

// ---------- kernels ----------

__global__ void cvt_f32_bf16(const float* __restrict__ s, __bf16* __restrict__ d, int n) {
  int i = blockIdx.x * blockDim.x + threadIdx.x;
  if (i < n) d[i] = f2bf(s[i]);
}

// ----- software-pipelined GEMM -----
// C = A(MxK) * W(NxK)^T ; 4 waves/block, each wave owns a 32x64 C-tile
// (block tile 128x64). Fragments double-buffered so loads for step k+1
// are in flight while the 8-wmma chain for step k issues.
struct Frags {
  v16bf a0, a1;          // rows row0, row0+16
  v16bf b0, b1, b2, b3;  // cols n0, n0+16, n0+32, n0+48
};

__device__ __forceinline__ void load_step(const __bf16* __restrict__ A,
                                          const __bf16* __restrict__ W,
                                          int Ktot, int row0, int n0, int k0,
                                          Frags& f) {
  f.a0 = load_frag_a(A, Ktot, row0,      k0);
  f.a1 = load_frag_a(A, Ktot, row0 + 16, k0);
  f.b0 = load_frag_b(W, Ktot, n0,        k0);
  f.b1 = load_frag_b(W, Ktot, n0 + 16,   k0);
  f.b2 = load_frag_b(W, Ktot, n0 + 32,   k0);
  f.b3 = load_frag_b(W, Ktot, n0 + 48,   k0);
}

__device__ __forceinline__ void mma_step(const Frags& f, v8f acc[2][4]) {
  acc[0][0] = wmma_bf16(f.a0, f.b0, acc[0][0]);
  acc[1][0] = wmma_bf16(f.a1, f.b0, acc[1][0]);
  acc[0][1] = wmma_bf16(f.a0, f.b1, acc[0][1]);
  acc[1][1] = wmma_bf16(f.a1, f.b1, acc[1][1]);
  acc[0][2] = wmma_bf16(f.a0, f.b2, acc[0][2]);
  acc[1][2] = wmma_bf16(f.a1, f.b2, acc[1][2]);
  acc[0][3] = wmma_bf16(f.a0, f.b3, acc[0][3]);
  acc[1][3] = wmma_bf16(f.a1, f.b3, acc[1][3]);
}

// mode 0: bf16 row-major out [M x N]
// mode 1: bf16 per-batch transposed out: Ct[(b*N + col)*MperB + (row - b*MperB)]
// mode 2: f32 row-major out + bias
__global__ __launch_bounds__(128)
void gemm_bf16_wmma(const __bf16* __restrict__ A, const __bf16* __restrict__ W,
                    const float* __restrict__ bias, void* __restrict__ Cout,
                    int M, int N, int Ktot, int mode, int MperB) {
  const int lane = threadIdx.x & 31;
  const int wv   = threadIdx.x >> 5;
  const int row0 = blockIdx.x * 128 + wv * 32;   // wave owns rows row0..row0+31
  const int n0   = blockIdx.y * 64;
  const int m16 = lane & 15, half = lane >> 4;

  v8f acc[2][4];
  const v8f vzero = {0.f,0.f,0.f,0.f,0.f,0.f,0.f,0.f};
#pragma unroll
  for (int i = 0; i < 2; ++i)
#pragma unroll
    for (int t = 0; t < 4; ++t) acc[i][t] = vzero;

  Frags fA, fB;
  load_step(A, W, Ktot, row0, n0, 0, fA);
  int k0 = 32;
  for (; k0 + 32 < Ktot; k0 += 64) {
    load_step(A, W, Ktot, row0, n0, k0, fB);        // in-flight during fA mmas
    mma_step(fA, acc);
    load_step(A, W, Ktot, row0, n0, k0 + 32, fA);   // in-flight during fB mmas
    mma_step(fB, acc);
  }
  if (k0 < Ktot) {
    load_step(A, W, Ktot, row0, n0, k0, fB);
    mma_step(fA, acc);
    mma_step(fB, acc);
  } else {
    mma_step(fA, acc);
  }

  if (mode == 2) {
    float* C = (float*)Cout;
#pragma unroll
    for (int t = 0; t < 4; ++t) {
      int col = n0 + 16 * t + m16;
      float bv = bias[col];
#pragma unroll
      for (int i = 0; i < 2; ++i)
#pragma unroll
        for (int r = 0; r < 8; ++r) {
          int row = row0 + 16 * i + r + 8 * half;
          C[(size_t)row * N + col] = acc[i][t][r] + bv;
        }
    }
  } else if (mode == 1) {
    __bf16* C = (__bf16*)Cout;
    int b = row0 / MperB;
#pragma unroll
    for (int t = 0; t < 4; ++t) {
      int col = n0 + 16 * t + m16;
#pragma unroll
      for (int i = 0; i < 2; ++i)
#pragma unroll
        for (int r = 0; r < 8; ++r) {
          int row = row0 + 16 * i + r + 8 * half;
          C[((size_t)b * N + col) * MperB + (row - b * MperB)] = f2bf(acc[i][t][r]);
        }
    }
  } else {
    __bf16* C = (__bf16*)Cout;
#pragma unroll
    for (int t = 0; t < 4; ++t) {
      int col = n0 + 16 * t + m16;
#pragma unroll
      for (int i = 0; i < 2; ++i)
#pragma unroll
        for (int r = 0; r < 8; ++r) {
          int row = row0 + 16 * i + r + 8 * half;
          C[(size_t)row * N + col] = f2bf(acc[i][t][r]);
        }
    }
  }
}

// ----- pipelined flash attention -----

__device__ __forceinline__ void load_kfrags(const __bf16* __restrict__ Kb,
                                            int key0, int hoff, v16bf kf[4]) {
  kf[0] = load_frag_b(Kb, DMODEL, key0,      hoff);
  kf[1] = load_frag_b(Kb, DMODEL, key0,      hoff + 32);
  kf[2] = load_frag_b(Kb, DMODEL, key0 + 16, hoff);
  kf[3] = load_frag_b(Kb, DMODEL, key0 + 16, hoff + 32);
}

__device__ __forceinline__ void attn_chunk(int key0, const v16bf& qf0, const v16bf& qf1,
                                           const v16bf kf[4],
                                           const __bf16* __restrict__ Vb, int hoff,
                                           float mr[8], float lr[8], v8f accO[4],
                                           __bf16* sP) {
  const int lane = threadIdx.x & 31;
  const int m16 = lane & 15, half = lane >> 4;
  const v8f vzero = {0.f,0.f,0.f,0.f,0.f,0.f,0.f,0.f};

  // S = Q * K^T  (two 16x16 score tiles; K-dim 64 = 2 x 32)
  v8f s0 = vzero, s1 = vzero;
  s0 = wmma_bf16(qf0, kf[0], s0);
  s0 = wmma_bf16(qf1, kf[1], s0);
  s1 = wmma_bf16(qf0, kf[2], s1);
  s1 = wmma_bf16(qf1, kf[3], s1);

  // issue V fragment loads now; the softmax VALU block below hides their latency
  v16bf vf0 = load_frag_b(Vb, MMK, hoff,      key0);
  v16bf vf1 = load_frag_b(Vb, MMK, hoff + 16, key0);
  v16bf vf2 = load_frag_b(Vb, MMK, hoff + 32, key0);
  v16bf vf3 = load_frag_b(Vb, MMK, hoff + 48, key0);

  // online softmax; row (r+8*half) lives in one VGPR across the 16 lanes
  // of a half -> reduce with xor 1,2,4,8.
#pragma unroll
  for (int r = 0; r < 8; ++r) {
    float v0 = s0[r] * SCALEF, v1 = s1[r] * SCALEF;
    float mx = fmaxf(v0, v1);
#pragma unroll
    for (int off = 1; off < 16; off <<= 1)
      mx = fmaxf(mx, __shfl_xor(mx, off, 32));
    float mn = fmaxf(mr[r], mx);
    float p0 = __expf(v0 - mn), p1 = __expf(v1 - mn);
    float rs = p0 + p1;
#pragma unroll
    for (int off = 1; off < 16; off <<= 1)
      rs += __shfl_xor(rs, off, 32);
    float alpha = __expf(mr[r] - mn);
    lr[r] = lr[r] * alpha + rs;
    mr[r] = mn;
#pragma unroll
    for (int t = 0; t < 4; ++t) accO[t][r] *= alpha;
    const int rw = r + 8 * half;
    sP[rw * 32 + m16]      = f2bf(p0);
    sP[rw * 32 + 16 + m16] = f2bf(p1);
  }
  __syncthreads();
  v16bf pf = load_frag_a_lds(sP);   // P as 16x32 A fragment
  __syncthreads();

  // O += P * V
  accO[0] = wmma_bf16(pf, vf0, accO[0]);
  accO[1] = wmma_bf16(pf, vf1, accO[1]);
  accO[2] = wmma_bf16(pf, vf2, accO[2]);
  accO[3] = wmma_bf16(pf, vf3, accO[3]);
}

// One wave per 16-row Q tile per (b,h). Q,K row-major [b][seq][h*64+d];
// Vt per-batch transposed [b][h*64+d][MMK].
__global__ __launch_bounds__(32)
void attn_wmma(const __bf16* __restrict__ Q, const __bf16* __restrict__ K,
               const __bf16* __restrict__ Vt, __bf16* __restrict__ O) {
  __shared__ __bf16 sP[16 * 32];

  const int lane = threadIdx.x & 31;
  const int m16 = lane & 15, half = lane >> 4;
  const int qt = blockIdx.x & 127;          // N/16 = 128 tiles
  const int h  = (blockIdx.x >> 7) & 15;
  const int b  = blockIdx.x >> 11;
  const int q0 = qt * 16;
  const int hoff = h * DH;

  const __bf16* Qb = Q  + (size_t)b * NN  * DMODEL;
  const __bf16* Kb = K  + (size_t)b * MMK * DMODEL;
  const __bf16* Vb = Vt + (size_t)b * DMODEL * MMK;

  v16bf qf0 = load_frag_a(Qb, DMODEL, q0, hoff);
  v16bf qf1 = load_frag_a(Qb, DMODEL, q0, hoff + 32);

  float mr[8], lr[8];
#pragma unroll
  for (int r = 0; r < 8; ++r) { mr[r] = -1e30f; lr[r] = 0.f; }
  const v8f vzero = {0.f,0.f,0.f,0.f,0.f,0.f,0.f,0.f};
  v8f accO[4];
#pragma unroll
  for (int t = 0; t < 4; ++t) accO[t] = vzero;

  const int NCH = MMK / 32;   // 64 chunks
  v16bf kA[4], kB[4];
  load_kfrags(Kb, 0, hoff, kA);

  int c = 0;
  for (; c + 2 < NCH; c += 2) {
    load_kfrags(Kb, (c + 1) * 32, hoff, kB);      // in flight during chunk c
    // prefetch next-next V tile rows (gfx1250 global_prefetch_b8)
    __builtin_prefetch(Vb + (size_t)(hoff + lane) * MMK + (c + 2) * 32, 0, 0);
    __builtin_prefetch(Vb + (size_t)(hoff + 32 + lane) * MMK + (c + 2) * 32, 0, 0);
    attn_chunk(c * 32, qf0, qf1, kA, Vb, hoff, mr, lr, accO, sP);
    load_kfrags(Kb, (c + 2) * 32, hoff, kA);      // in flight during chunk c+1
    attn_chunk((c + 1) * 32, qf0, qf1, kB, Vb, hoff, mr, lr, accO, sP);
  }
  // tail: chunks c (in kA) and c+1
  load_kfrags(Kb, (c + 1) * 32, hoff, kB);
  attn_chunk(c * 32, qf0, qf1, kA, Vb, hoff, mr, lr, accO, sP);
  attn_chunk((c + 1) * 32, qf0, qf1, kB, Vb, hoff, mr, lr, accO, sP);

  // epilogue: normalize and store bf16 row-major [b][n][h*64+d]
#pragma unroll
  for (int r = 0; r < 8; ++r) {
    float inv = 1.0f / lr[r];
    int row = q0 + r + 8 * half;
#pragma unroll
    for (int t = 0; t < 4; ++t) {
      O[((size_t)b * NN + row) * DMODEL + hoff + 16 * t + m16] = f2bf(accO[t][r] * inv);
    }
  }
}

// ---------- host ----------

extern "C" void kernel_launch(void* const* d_in, const int* in_sizes, int n_in,
                              void* d_out, int out_size, void* d_ws, size_t ws_size,
                              hipStream_t stream) {
  (void)in_sizes; (void)n_in; (void)out_size; (void)ws_size;
  const float* x   = (const float*)d_in[0];  // [2,2048,1024]
  const float* ctx = (const float*)d_in[1];  // [2,2048,1024]
  const float* Wq  = (const float*)d_in[2];  // [1024,1024]
  const float* Wk  = (const float*)d_in[3];
  const float* Wv  = (const float*)d_in[4];
  const float* Wo  = (const float*)d_in[5];
  const float* bo  = (const float*)d_in[6];  // [1024]
  float* out = (float*)d_out;                // [2,2048,1024] f32

  const size_t SX = (size_t)BB * NN * QD;      // 4M elems
  const size_t SW = (size_t)DMODEL * QD;       // 1M elems

  char* w = (char*)d_ws;
  __bf16* xb = (__bf16*)w;  w += SX * 2;
  __bf16* cb = (__bf16*)w;  w += SX * 2;
  __bf16* wq = (__bf16*)w;  w += SW * 2;
  __bf16* wk = (__bf16*)w;  w += SW * 2;
  __bf16* wv = (__bf16*)w;  w += SW * 2;
  __bf16* wo = (__bf16*)w;  w += SW * 2;
  __bf16* Qb = (__bf16*)w;  w += SX * 2;
  __bf16* Kb = (__bf16*)w;  w += SX * 2;
  __bf16* Vt = (__bf16*)w;  w += SX * 2;
  __bf16* Ob = (__bf16*)w;  w += SX * 2;   // total ~56 MB

  cvt_f32_bf16<<<(unsigned)((SX + 255) / 256), 256, 0, stream>>>(x,   xb, (int)SX);
  cvt_f32_bf16<<<(unsigned)((SX + 255) / 256), 256, 0, stream>>>(ctx, cb, (int)SX);
  cvt_f32_bf16<<<(unsigned)((SW + 255) / 256), 256, 0, stream>>>(Wq,  wq, (int)SW);
  cvt_f32_bf16<<<(unsigned)((SW + 255) / 256), 256, 0, stream>>>(Wk,  wk, (int)SW);
  cvt_f32_bf16<<<(unsigned)((SW + 255) / 256), 256, 0, stream>>>(Wv,  wv, (int)SW);
  cvt_f32_bf16<<<(unsigned)((SW + 255) / 256), 256, 0, stream>>>(Wo,  wo, (int)SW);

  const int Mrows = BB * NN;   // 4096
  dim3 blk(128, 1, 1);
  dim3 g1(Mrows / 128, DMODEL / 64, 1);
  // Q = x @ Wq^T (bf16 row-major)
  gemm_bf16_wmma<<<g1, blk, 0, stream>>>(xb, wq, nullptr, (void*)Qb, Mrows, DMODEL, QD, 0, MMK);
  // K = ctx @ Wk^T (bf16 row-major)
  gemm_bf16_wmma<<<g1, blk, 0, stream>>>(cb, wk, nullptr, (void*)Kb, Mrows, DMODEL, QD, 0, MMK);
  // V = ctx @ Wv^T, stored per-batch transposed [b][dmodel][M]
  gemm_bf16_wmma<<<g1, blk, 0, stream>>>(cb, wv, nullptr, (void*)Vt, Mrows, DMODEL, QD, 1, MMK);

  // attention: one wave per 16-row q tile per (b,h)
  attn_wmma<<<dim3(BB * HH * (NN / 16), 1, 1), dim3(32, 1, 1), 0, stream>>>(Qb, Kb, Vt, Ob);

  // out = O @ Wo^T + bo (f32)
  gemm_bf16_wmma<<<g1, blk, 0, stream>>>(Ob, wo, bo, (void*)out, Mrows, DMODEL, QD, 2, MMK);
}
